// CodeGraphEnc_49692771614969
// MI455X (gfx1250) — matrix operand order repaired and checked by reading, not compiled, not run on hardware
//
#include <hip/hip_runtime.h>
#include <hip/hip_bf16.h>

// ---------------------------------------------------------------------------
// CodeGraphEnc for MI455X (gfx1250, wave32, WMMA).
// Memory-bound workload (~19 GB aggregation traffic @ 23.3 TB/s); GEMMs
// (722 GFLOP) done with v_wmma_f32_16x16x32_bf16, weights L2-resident.
// ---------------------------------------------------------------------------

#define NN   131072
#define BB   64
#define DD   256
#define LL   6
#define RR   6
#define EE   262144
#define GMX  2048
#define CMX  1024

typedef __attribute__((ext_vector_type(16))) __bf16        v16bf;
typedef __attribute__((ext_vector_type(8)))  float         v8f;
typedef __attribute__((ext_vector_type(4)))  unsigned int  v4u;

union BfVec { v16bf v; v4u u[2]; };

__device__ __forceinline__ unsigned short f2bf(float f) {
    unsigned int u = __float_as_uint(f);
    u += 0x7FFFu + ((u >> 16) & 1u);          // round-to-nearest-even
    return (unsigned short)(u >> 16);
}

// ---------------- embedding gather: x[n][:] = emb[ids[n]][:] ---------------
__global__ __launch_bounds__(256) void gather_embed(
    const float* __restrict__ emb, const int* __restrict__ ids,
    float* __restrict__ x)
{
    int node = blockIdx.x * 8 + (threadIdx.x >> 5);
    int lane = threadIdx.x & 31;
    const float4* s = (const float4*)(emb + (size_t)ids[node] * DD) + lane;
    float4*       d = (float4*)(x + (size_t)node * DD) + lane;
    d[0]  = s[0];
    d[32] = s[32];
}

// ---------------- f32 -> bf16 (optional per-row mean scale) ----------------
__global__ __launch_bounds__(256) void cvt_bf16(
    const float* __restrict__ in, const float* __restrict__ rowscale,
    unsigned short* __restrict__ out, size_t n)
{
    size_t i = (size_t)blockIdx.x * 256 + threadIdx.x;
    if (i >= n) return;
    float v = in[i];
    if (rowscale) v *= rowscale[i >> 8];      // row = i / DD
    out[i] = f2bf(v);
}

// ---------------- in-degree counting + reciprocal --------------------------
__global__ __launch_bounds__(256) void edge_count(
    const int* __restrict__ edges, int* __restrict__ cnt)
{
    size_t i = (size_t)blockIdx.x * 256 + threadIdx.x;   // over R*E
    int r = (int)(i / EE), e = (int)(i % EE);
    int d = edges[(size_t)r * 2 * EE + EE + e];
    atomicAdd(&cnt[(size_t)r * NN + d], 1);
}

__global__ __launch_bounds__(256) void recip_counts(int* __restrict__ cnt)
{
    size_t i = (size_t)blockIdx.x * 256 + threadIdx.x;
    if (i >= (size_t)RR * NN) return;
    int c = cnt[i];
    ((float*)cnt)[i] = 1.0f / (float)(c > 0 ? c : 1);
}

// ---------------- edge scatter-add (one wave32 per edge) -------------------
__global__ __launch_bounds__(256) void scatter_edges(
    const int* __restrict__ src, const int* __restrict__ dst,
    const float* __restrict__ x, float* __restrict__ agg)
{
    int e    = (int)(((size_t)blockIdx.x * 256 + threadIdx.x) >> 5);
    int lane = threadIdx.x & 31;
    if (e >= EE) return;
    int s = src[e], d = dst[e];
    const float4* xr = (const float4*)(x + (size_t)s * DD);
    float4 a = xr[lane];
    float4 b = xr[lane + 32];
    float* ad = agg + (size_t)d * DD;
    int o = lane * 4;
    atomicAdd(ad + o + 0,   a.x); atomicAdd(ad + o + 1,   a.y);
    atomicAdd(ad + o + 2,   a.z); atomicAdd(ad + o + 3,   a.w);
    atomicAdd(ad + o + 128, b.x); atomicAdd(ad + o + 129, b.y);
    atomicAdd(ad + o + 130, b.z); atomicAdd(ad + o + 131, b.w);
}

// ---------------- bf16 WMMA GEMM: Y (+)= A[N,256] . W[256,256]^T -----------
// Block: 256 threads = 8 waves; 128 rows x 64 cols per block.
// Wave w -> M-tile m0 = blockIdx.x*128 + 16w; 4 N-tiles of 16.
// A/B lane layouts follow CDNA5 ISA 16-bit WMMA VGPR striping exactly.
__global__ __launch_bounds__(256) void gemm_bf16_acc(
    const unsigned short* __restrict__ A,   // [NN, DD] bf16
    const unsigned short* __restrict__ W,   // [DD out, DD k] bf16 (native layout)
    float* __restrict__ Y, int init)
{
    const int lane = threadIdx.x & 31;
    const int wave = threadIdx.x >> 5;
    const int half = lane >> 4;              // 0: lanes 0-15, 1: lanes 16-31
    const int l16  = lane & 15;
    const int m0   = blockIdx.x * 128 + wave * 16;
    const int n0   = blockIdx.y * 64;

    v8f acc[4];
    if (init) {
#pragma unroll
        for (int j = 0; j < 4; ++j)
#pragma unroll
            for (int i = 0; i < 8; ++i) acc[j][i] = 0.0f;
    } else {
#pragma unroll
        for (int j = 0; j < 4; ++j)
#pragma unroll
            for (int i = 0; i < 8; ++i)
                acc[j][i] = Y[(size_t)(m0 + i + 8 * half) * DD + n0 + 16 * j + l16];
    }

    const unsigned short* arow = A + (size_t)(m0 + l16) * DD;
#pragma unroll
    for (int k0 = 0; k0 < DD; k0 += 32) {
        BfVec a;
        // A 16x32 bf16: lanes<16 K{k0..k0+7, k0+16..k0+23}; lanes>=16 shifted +8
        a.u[0] = *(const v4u*)(arow + k0 + 8 * half);
        a.u[1] = *(const v4u*)(arow + k0 + 8 * half + 16);
        if (k0 + 32 < DD) __builtin_prefetch(arow + k0 + 32, 0, 1); // global_prefetch_b8
#pragma unroll
        for (int j = 0; j < 4; ++j) {
            // B 32x16 bf16: lanes<16 K{k0..k0+7, k0+16..k0+23}; lanes>=16 shifted +8
            const unsigned short* wrow =
                W + (size_t)(n0 + 16 * j + l16) * DD + k0 + 8 * half;
            BfVec b;
            b.u[0] = *(const v4u*)(wrow);
            b.u[1] = *(const v4u*)(wrow + 16);
            acc[j] = __builtin_amdgcn_wmma_f32_16x16x32_bf16(
                false, a.v, false, b.v, (short)0, acc[j], false, false);
        }
    }

#pragma unroll
    for (int j = 0; j < 4; ++j)
#pragma unroll
        for (int i = 0; i < 8; ++i)
            Y[(size_t)(m0 + i + 8 * half) * DD + n0 + 16 * j + l16] = acc[j][i];
}

// ---------------- h = x + relu(y + sum_r bl[r][d]) (into y) ----------------
__global__ __launch_bounds__(256) void add_relu(
    const float* __restrict__ x, float* __restrict__ y,
    const float* __restrict__ bl_l)  // [RR, DD]
{
    size_t i = (size_t)blockIdx.x * 256 + threadIdx.x;
    if (i >= (size_t)NN * DD) return;
    int d = (int)(i & 255);
    float bias = 0.0f;
#pragma unroll
    for (int r = 0; r < RR; ++r) bias += bl_l[r * DD + d];
    float v = y[i] + bias;
    v = v > 0.0f ? v : 0.0f;
    y[i] = x[i] + v;
}

// ---------------- GraphNorm: stats pass (atomic partials) ------------------
__global__ __launch_bounds__(256) void gn_stats(
    const float* __restrict__ h, float* __restrict__ stats) // [BB][2][DD]
{
    int g = blockIdx.x >> 4;          // 16 row-chunks of 128 per graph
    int c = blockIdx.x & 15;
    int d = threadIdx.x;
    const float* hp = h + ((size_t)g * GMX + (size_t)c * 128) * DD + d;
    float s = 0.0f, s2 = 0.0f;
    for (int r = 0; r < 128; ++r) { float v = hp[(size_t)r * DD]; s += v; s2 += v * v; }
    atomicAdd(&stats[(size_t)g * 2 * DD + d], s);
    atomicAdd(&stats[(size_t)g * 2 * DD + DD + d], s2);
}

// ---------------- GraphNorm: apply (var = E[x^2] - mu^2*(2a - a^2)) --------
__global__ __launch_bounds__(256) void gn_apply(
    const float* __restrict__ h, const float* __restrict__ stats,
    const float* __restrict__ w, const float* __restrict__ b,
    const float* __restrict__ al, float* __restrict__ out)
{
    size_t i = (size_t)blockIdx.x * 256 + threadIdx.x;
    if (i >= (size_t)NN * DD) return;
    int d = (int)(i & 255);
    int g = (int)((i >> 8) >> 11);
    const float inv_n = 1.0f / (float)GMX;
    float mu  = stats[(size_t)g * 2 * DD + d] * inv_n;
    float ex2 = stats[(size_t)g * 2 * DD + DD + d] * inv_n;
    float a   = al[d];
    float var = ex2 - mu * mu * (2.0f * a - a * a);
    float rinv = rsqrtf(var + 1e-5f);
    out[i] = w[d] * (h[i] - a * mu) * rinv + b[d];
}

// ---------------- per-graph code-node compaction (wave32 ballot scan) ------
__global__ void code_positions(const unsigned char* __restrict__ mask,
                               int* __restrict__ cpos)
{
    int g = blockIdx.x, lane = threadIdx.x;   // 32 threads per graph
    int running = 0;
    for (int c = 0; c < GMX / 32; ++c) {
        int n = g * GMX + c * 32 + lane;
        int m = mask[n] ? 1 : 0;
        unsigned int bal = (unsigned int)__ballot(m);   // wave32: low 32 bits
        int pos = running + __popc(bal & ((1u << lane) - 1u));
        cpos[n] = (m && pos < CMX) ? pos : -1;
        running += __popc(bal);
    }
}

__global__ __launch_bounds__(256) void code_scatter(
    const float* __restrict__ x, const int* __restrict__ cpos,
    const int* __restrict__ smap, float* __restrict__ out_code,
    int* __restrict__ out_map)
{
    int n    = (int)(((size_t)blockIdx.x * 256 + threadIdx.x) >> 5);
    int lane = threadIdx.x & 31;
    if (n >= NN) return;
    int cp = cpos[n];
    if (cp < 0) return;
    int g = n >> 11;
    const float4* s = (const float4*)(x + (size_t)n * DD) + lane;
    float4* d = (float4*)(out_code + ((size_t)g * CMX + cp) * DD) + lane;
    d[0]  = s[0];
    d[32] = s[32];
    if (lane == 0) out_map[(size_t)g * CMX + cp] = smap[n];
}

// ---------------------------------------------------------------------------
extern "C" void kernel_launch(void* const* d_in, const int* in_sizes, int n_in,
                              void* d_out, int out_size, void* d_ws, size_t ws_size,
                              hipStream_t stream)
{
    (void)in_sizes; (void)n_in; (void)out_size;

    const int*           node_ids = (const int*)d_in[0];
    const int*           src_map  = (const int*)d_in[1];
    const unsigned char* cmask    = (const unsigned char*)d_in[2];
    const int*           edges    = (const int*)d_in[4];   // [R,2,E]
    const float*         emb      = (const float*)d_in[5];
    const float*         Wl       = (const float*)d_in[6]; // [L,R,D,D]
    const float*         bl       = (const float*)d_in[7]; // [L,R,D]
    const float*         Wr       = (const float*)d_in[8]; // [L,D,D]
    const float*         gnw      = (const float*)d_in[9];
    const float*         gnb      = (const float*)d_in[10];
    const float*         gna      = (const float*)d_in[11];

    // ---- workspace layout (bytes) ----
    char* ws = (char*)d_ws;
    const size_t XB   = (size_t)NN * DD * 4;
    size_t off = 0;
    float*          xbuf  = (float*)(ws + off);          off += XB;   // x
    float*          ybuf  = (float*)(ws + off);          off += XB;   // y / h
    float*          agg   = (float*)(ws + off);          off += XB;   // one relation agg
    unsigned short* xh    = (unsigned short*)(ws + off); off += XB / 2;
    unsigned short* aggh  = (unsigned short*)(ws + off); off += XB / 2;
    unsigned short* wlh   = (unsigned short*)(ws + off); off += (size_t)LL * RR * DD * DD * 2;
    unsigned short* wrh   = (unsigned short*)(ws + off); off += (size_t)LL * DD * DD * 2;
    int*            cnt   = (int*)(ws + off);            off += (size_t)RR * NN * 4; // -> rcnt
    int*            cpos  = (int*)(ws + off);            off += (size_t)NN * 4;
    float*          stats = (float*)(ws + off);          off += (size_t)BB * 2 * DD * 4;
    (void)ws_size;

    const dim3 B256(256);

    // 1) embeddings
    gather_embed<<<NN / 8, B256, 0, stream>>>(emb, node_ids, xbuf);

    // 2) bf16 weights (native [d_out][k] layout — no transpose needed for x@W^T)
    {
        size_t nWl = (size_t)LL * RR * DD * DD;
        size_t nWr = (size_t)LL * DD * DD;
        cvt_bf16<<<(unsigned)((nWl + 255) / 256), B256, 0, stream>>>(Wl, nullptr, wlh, nWl);
        cvt_bf16<<<(unsigned)((nWr + 255) / 256), B256, 0, stream>>>(Wr, nullptr, wrh, nWr);
    }

    // 3) in-degree counts -> reciprocals (in place)
    hipMemsetAsync(cnt, 0, (size_t)RR * NN * 4, stream);
    edge_count<<<RR * EE / 256, B256, 0, stream>>>(edges, cnt);
    recip_counts<<<(RR * NN) / 256, B256, 0, stream>>>(cnt);
    const float* rcnt = (const float*)cnt;

    const dim3 ggrid(NN / 128, DD / 64);    // 1024 x 4 WMMA blocks
    const size_t nXD = (size_t)NN * DD;

    // 4) layers
    for (int l = 0; l < LL; ++l) {
        cvt_bf16<<<(unsigned)(nXD / 256), B256, 0, stream>>>(xbuf, nullptr, xh, nXD);

        // root term: y = x @ Wr[l]^T
        gemm_bf16_acc<<<ggrid, B256, 0, stream>>>(
            xh, wrh + (size_t)l * DD * DD, ybuf, 1);

        for (int r = 0; r < RR; ++r) {
            hipMemsetAsync(agg, 0, XB, stream);
            scatter_edges<<<EE * 32 / 256, B256, 0, stream>>>(
                edges + (size_t)r * 2 * EE,             // src
                edges + (size_t)r * 2 * EE + EE,        // dst
                xbuf, agg);
            cvt_bf16<<<(unsigned)(nXD / 256), B256, 0, stream>>>(
                agg, rcnt + (size_t)r * NN, aggh, nXD); // mean = sum * 1/cnt
            gemm_bf16_acc<<<ggrid, B256, 0, stream>>>(
                aggh, wlh + ((size_t)l * RR + r) * DD * DD, ybuf, 0);
        }

        // h = x + relu(y + bias), stored into ybuf
        add_relu<<<(unsigned)(nXD / 256), B256, 0, stream>>>(
            xbuf, ybuf, bl + (size_t)l * RR * DD);

        // GraphNorm(h) -> xbuf (next layer input)
        hipMemsetAsync(stats, 0, (size_t)BB * 2 * DD * 4, stream);
        gn_stats<<<BB * 16, B256, 0, stream>>>(ybuf, stats);
        gn_apply<<<(unsigned)(nXD / 256), B256, 0, stream>>>(
            ybuf, stats, gnw + l * DD, gnb + l * DD, gna + l * DD, xbuf);
    }

    // 5) outputs
    float* out0 = (float*)d_out;                       // [B, GMX, D] == x (dense batch)
    float* out1 = out0 + (size_t)BB * GMX * DD;        // [B, CMX, D]
    int*   out2 = (int*)(out1 + (size_t)BB * CMX * DD);// [B, CMX] (int32 bits)

    hipMemcpyAsync(out0, xbuf, XB, hipMemcpyDeviceToDevice, stream);
    hipMemsetAsync(out1, 0, (size_t)BB * CMX * DD * 4, stream);
    hipMemsetAsync(out2, 0, (size_t)BB * CMX * 4, stream);

    code_positions<<<BB, 32, 0, stream>>>(cmask, cpos);
    code_scatter<<<NN * 32 / 256, B256, 0, stream>>>(xbuf, cpos, src_map, out1, out2);
}
